// DeltaNetBlock_45397804319522
// MI455X (gfx1250) — matrix operand order repaired and checked by reading, not compiled, not run on hardware
//
#include <hip/hip_runtime.h>

// ---------------------------------------------------------------------------
// DeltaNet block for gfx1250 (MI455X):
//   - one streaming pass converts all fp32 weights -> bf16 (103MB, L2-resident)
//   - GEMMs: bf16 WMMA 16x16x32, two-stage software pipeline over
//     double-buffered LDS staged with global_load_async_to_lds_b128 +
//     statically-placed s_wait_asynccnt (CDNA5 async path, in-order completion)
//   - delta-rule scan: sequential VALU/LDS, one head per block
// Workspace assumption: ws_size >= ~195 MB. Requires K % 64 == 0, K >= 128.
// ---------------------------------------------------------------------------

#define HID 2048
#define TSEQ 1024
#define NHEAD 32
#define HDIM 64
#define INTER_DIM 5632
#define EPSF 1e-6f

typedef __bf16 bf16_t;
typedef __attribute__((ext_vector_type(16))) __bf16 v16bf;
typedef __attribute__((ext_vector_type(8))) float v8f;

#if defined(__has_builtin)
#define HAS_B(x) __has_builtin(x)
#else
#define HAS_B(x) 0
#endif

__device__ __forceinline__ float silu1(float x) { return x / (1.0f + __expf(-x)); }

// Low 32 bits of a generic LDS pointer == LDS byte offset (ISA 10.2: aperture
// translation keeps addr[31:0]); that is what async-load VDST expects.
__device__ __forceinline__ uint32_t lds_off32(const void* p) {
  return (uint32_t)(uintptr_t)p;
}
// CDNA5 async copy: 16 bytes per lane, global -> LDS, tracked by ASYNCcnt.
__device__ __forceinline__ void async_cp_b128(uint32_t lds_byte_off, const void* gsrc) {
  asm volatile("global_load_async_to_lds_b128 %0, %1, off"
               :: "v"(lds_byte_off), "v"(gsrc) : "memory");
}
__device__ __forceinline__ void wait_async_le4() {
#if HAS_B(__builtin_amdgcn_s_wait_asynccnt)
  __builtin_amdgcn_s_wait_asynccnt(4);
#else
  asm volatile("s_wait_asynccnt 0x4" ::: "memory");
#endif
}
__device__ __forceinline__ void wait_async_0() {
#if HAS_B(__builtin_amdgcn_s_wait_asynccnt)
  __builtin_amdgcn_s_wait_asynccnt(0);
#else
  asm volatile("s_wait_asynccnt 0x0" ::: "memory");
#endif
}

// ---------------- fp32 -> bf16 packed convert (8 elems/thread) ----------------
__global__ __launch_bounds__(256)
void convert_f32_bf16_kernel(const float* __restrict__ src, bf16_t* __restrict__ dst,
                             long n) {
  const long i = ((long)blockIdx.x * 256 + threadIdx.x) * 8;
  if (i >= n) return;
  float4 a = *(const float4*)(src + i);
  float4 b = *(const float4*)(src + i + 4);
  union { bf16_t h[8]; float4 f; } u;
  u.h[0] = (bf16_t)a.x; u.h[1] = (bf16_t)a.y; u.h[2] = (bf16_t)a.z; u.h[3] = (bf16_t)a.w;
  u.h[4] = (bf16_t)b.x; u.h[5] = (bf16_t)b.y; u.h[6] = (bf16_t)b.z; u.h[7] = (bf16_t)b.w;
  *(float4*)(dst + i) = u.f;
}

// ---------------- RMSNorm: fp32 row -> bf16 row ----------------
__global__ __launch_bounds__(256)
void rmsnorm_bf16_kernel(const float* __restrict__ x,
                         const float* __restrict__ w,
                         bf16_t* __restrict__ out) {
  const int row = blockIdx.x;
  const float* xr = x + (size_t)row * HID;
  __shared__ float red[8];
  float s = 0.f;
  for (int c = threadIdx.x; c < HID; c += 256) { float v = xr[c]; s += v * v; }
#pragma unroll
  for (int off = 16; off > 0; off >>= 1) s += __shfl_down(s, off);
  if ((threadIdx.x & 31) == 0) red[threadIdx.x >> 5] = s;
  __syncthreads();
  if (threadIdx.x == 0) {
    float t = 0.f;
#pragma unroll
    for (int i = 0; i < 8; ++i) t += red[i];
    red[0] = rsqrtf(t * (1.0f / (float)HID) + EPSF);
  }
  __syncthreads();
  const float r = red[0];
  bf16_t* orow = out + (size_t)row * HID;
  for (int c = threadIdx.x; c < HID; c += 256) orow[c] = (bf16_t)(xr[c] * r * w[c]);
}

// ---------------- WMMA GEMM: C[M,N] = A_bf16[M,K] * B_bf16[N,K]^T ----------------
// Block 256 thr = 8 waves; block tile 128x128; wave tile 32x64 (2x4 frags,
// 8 v_wmma per K-step); BK=32; two-stage pipeline, statically unrolled.
#define BM 128
#define BN 128
#define BK 32
#define LDP (BK + 8)  // row pad: rows stay 16B-aligned, fewer bank conflicts

// A-matrix 16-bit fragment (16x32): kb=(lane>>4)*8;
// elems 0..7 -> K=kb..kb+7, elems 8..15 -> K=kb+16..kb+23.
__device__ __forceinline__ v16bf load_fragA(const bf16_t* row, int kb) {
  union { v16bf v; float4 h[2]; } u;
  u.h[0] = *(const float4*)(row + kb);
  u.h[1] = *(const float4*)(row + kb + 16);
  return u.v;
}
// B-matrix 16-bit fragment (32x16): kb2=(lane>>4)*8; elems 0..7 -> K=kb2..kb2+7, elems 8..15 -> K=kb2+16..kb2+23.
__device__ __forceinline__ v16bf load_fragB(const bf16_t* row, int kb2) {
  union { v16bf v; float4 h[2]; } u;
  u.h[0] = *(const float4*)(row + kb2);
  u.h[1] = *(const float4*)(row + kb2 + 16);
  return u.v;
}

__global__ __launch_bounds__(256)
void gemm_bf16_wmma_async_kernel(const bf16_t* __restrict__ A,
                                 const bf16_t* __restrict__ B,
                                 float* __restrict__ C,
                                 int M, int N, int K) {
  __shared__ bf16_t As[2][BM][LDP];
  __shared__ bf16_t Bs[2][BN][LDP];

  const int bm = blockIdx.y * BM;
  const int bn = blockIdx.x * BN;
  const int tid = threadIdx.x;
  const int wave = tid >> 5;
  const int lane = tid & 31;
  const int wm = (wave >> 1) * 32;   // wave M origin: 0,32,64,96
  const int wn = (wave & 1) * 64;    // wave N origin: 0,64
  const int rsel = lane & 15;
  const int kbA = (lane >> 4) * 8;
  const int kbB = (lane >> 4) * 8;

  // staging: thread -> row=tid>>1, col=(tid&1)*16 (16 bf16 = 2 x b128)
  const int sr = tid >> 1;
  const int sc = (tid & 1) * 16;
  const bf16_t* agk = A + (size_t)(bm + sr) * K + sc;
  const int brow = (bn + sr < N) ? (bn + sr) : (N - 1);  // clamp: keeps issue
  const bf16_t* bgk = B + (size_t)brow * K + sc;         // count wave-uniform

  const uint32_t aoff[2] = { lds_off32(&As[0][sr][sc]), lds_off32(&As[1][sr][sc]) };
  const uint32_t boff[2] = { lds_off32(&Bs[0][sr][sc]), lds_off32(&Bs[1][sr][sc]) };

  auto issue = [&](int b, const bf16_t* ap, const bf16_t* bp) {
    async_cp_b128(aoff[b],      ap);
    async_cp_b128(aoff[b] + 16, ap + 8);
    async_cp_b128(boff[b],      bp);
    async_cp_b128(boff[b] + 16, bp + 8);
  };

  v8f acc[2][4] = {};
  auto compute = [&](int b) {
    v16bf afr0 = load_fragA(&As[b][wm + rsel][0], kbA);
    v16bf afr1 = load_fragA(&As[b][wm + 16 + rsel][0], kbA);
    v16bf bfr0 = load_fragB(&Bs[b][wn + rsel][0], kbB);
    v16bf bfr1 = load_fragB(&Bs[b][wn + 16 + rsel][0], kbB);
    v16bf bfr2 = load_fragB(&Bs[b][wn + 32 + rsel][0], kbB);
    v16bf bfr3 = load_fragB(&Bs[b][wn + 48 + rsel][0], kbB);
    acc[0][0] = __builtin_amdgcn_wmma_f32_16x16x32_bf16(false, afr0, false, bfr0,
                                                        (short)0, acc[0][0], false, false);
    acc[0][1] = __builtin_amdgcn_wmma_f32_16x16x32_bf16(false, afr0, false, bfr1,
                                                        (short)0, acc[0][1], false, false);
    acc[0][2] = __builtin_amdgcn_wmma_f32_16x16x32_bf16(false, afr0, false, bfr2,
                                                        (short)0, acc[0][2], false, false);
    acc[0][3] = __builtin_amdgcn_wmma_f32_16x16x32_bf16(false, afr0, false, bfr3,
                                                        (short)0, acc[0][3], false, false);
    acc[1][0] = __builtin_amdgcn_wmma_f32_16x16x32_bf16(false, afr1, false, bfr0,
                                                        (short)0, acc[1][0], false, false);
    acc[1][1] = __builtin_amdgcn_wmma_f32_16x16x32_bf16(false, afr1, false, bfr1,
                                                        (short)0, acc[1][1], false, false);
    acc[1][2] = __builtin_amdgcn_wmma_f32_16x16x32_bf16(false, afr1, false, bfr2,
                                                        (short)0, acc[1][2], false, false);
    acc[1][3] = __builtin_amdgcn_wmma_f32_16x16x32_bf16(false, afr1, false, bfr3,
                                                        (short)0, acc[1][3], false, false);
  };

  // Two-stage pipeline, statically unrolled by 64 along K (K%64==0, K>=128).
  // Invariant at loop head: buf0 holds tile@k0 (4 oldest async ops),
  // buf1 holds tile@k0+32 (4 newest). Async loads complete in order.
  issue(0, agk, bgk);
  issue(1, agk + BK, bgk + BK);
  agk += 2 * BK; bgk += 2 * BK;
  int k0 = 0;
  for (; k0 + 64 < K; k0 += 64) {
    wait_async_le4();            // buf0 tile landed
    __syncthreads();
    compute(0);
    __syncthreads();             // all waves done reading buf0
    issue(0, agk, bgk);          // refill buf0 with tile@k0+64
    wait_async_le4();            // buf1 tile landed
    __syncthreads();
    compute(1);
    __syncthreads();             // all waves done reading buf1
    issue(1, agk + BK, bgk + BK);  // refill buf1 with tile@k0+96
    agk += 64; bgk += 64;
  }
  // tail: buf0@K-64, buf1@K-32 still in flight
  wait_async_le4();
  __syncthreads();
  compute(0);
  __syncthreads();
  wait_async_0();
  __syncthreads();
  compute(1);

  // C fragment: VGPR r -> M = r + (lane>=16)*8 ; N = lane&15
  const int mh = (lane >> 4) * 8;
  const int nsel = lane & 15;
#pragma unroll
  for (int mi = 0; mi < 2; ++mi) {
#pragma unroll
    for (int ni = 0; ni < 4; ++ni) {
      const int n = bn + wn + ni * 16 + nsel;
      if (n < N) {
        const int mbase = bm + wm + mi * 16 + mh;
#pragma unroll
        for (int r = 0; r < 8; ++r)
          C[(size_t)(mbase + r) * N + n] = acc[mi][ni][r];
      }
    }
  }
}

// ---------------- causal depthwise conv (K=4) + double SiLU + per-head L2 ----------------
__global__ __launch_bounds__(64)
void conv_silu_l2_kernel(const float* __restrict__ qpre, const float* __restrict__ kpre,
                         const float* __restrict__ vpre,
                         const float* __restrict__ wq, const float* __restrict__ wk,
                         const float* __restrict__ wv,
                         float* __restrict__ qo, float* __restrict__ ko,
                         float* __restrict__ vo) {
  const int t = blockIdx.x;
  const int head = blockIdx.y;
  const int d = threadIdx.x;
  const int c = head * HDIM + d;
  float aq = 0.f, ak = 0.f, av = 0.f;
#pragma unroll
  for (int j = 0; j < 4; ++j) {
    const int tt = t - 3 + j;
    if (tt >= 0) {
      const size_t idx = (size_t)tt * HID + c;
      aq += qpre[idx] * wq[c * 4 + j];
      ak += kpre[idx] * wk[c * 4 + j];
      av += vpre[idx] * wv[c * 4 + j];
    }
  }
  float q = silu1(silu1(aq));
  float k = silu1(silu1(ak));
  float v = silu1(silu1(av));
  __shared__ float sq[HDIM], sk[HDIM];
  sq[d] = q * q; sk[d] = k * k;
  __syncthreads();
#pragma unroll
  for (int off = 32; off > 0; off >>= 1) {
    if (d < off) { sq[d] += sq[d + off]; sk[d] += sk[d + off]; }
    __syncthreads();
  }
  const float qn = fmaxf(sqrtf(sq[0]), 1e-12f);
  const float kn = fmaxf(sqrtf(sk[0]), 1e-12f);
  const size_t o = (size_t)t * HID + c;
  qo[o] = q / qn; ko[o] = k / kn; vo[o] = v;
}

// ---------------- sequential delta-rule scan: one head per block ----------------
__global__ __launch_bounds__(64)
void delta_scan_kernel(const float* __restrict__ q, const float* __restrict__ k,
                       const float* __restrict__ v, const float* __restrict__ beta,
                       float* __restrict__ o, float* __restrict__ S_out) {
  const int head = blockIdx.x;
  const int d = threadIdx.x;   // row of state S (v index)
  float S[HDIM];
#pragma unroll
  for (int j = 0; j < HDIM; ++j) S[j] = 0.f;
  __shared__ float ks[HDIM], qs[HDIM];
  for (int t = 0; t < TSEQ; ++t) {
    const size_t base = (size_t)t * HID + head * HDIM;
    ks[d] = k[base + d];
    qs[d] = q[base + d];
    const float vt = v[base + d];
    const float bt = beta[t * NHEAD + head];
    __syncthreads();
    float err = -vt;
#pragma unroll 16
    for (int j = 0; j < HDIM; ++j) err += S[j] * ks[j];
    const float be = bt * err;
    float ob = 0.f;
#pragma unroll 16
    for (int j = 0; j < HDIM; ++j) { S[j] -= be * ks[j]; ob += S[j] * qs[j]; }
    o[base + d] = ob;
    __syncthreads();
  }
  float* srow = S_out + (size_t)head * HDIM * HDIM + (size_t)d * HDIM;
#pragma unroll 16
  for (int j = 0; j < HDIM; ++j) srow[j] = S[j];
}

// ---------------- elementwise helpers ----------------
__global__ void sigmoid_inplace_kernel(float* __restrict__ p, int n) {
  const int i = blockIdx.x * 256 + threadIdx.x;
  if (i < n) p[i] = 1.0f / (1.0f + __expf(-p[i]));
}
__global__ void add_kernel(const float* __restrict__ a, const float* __restrict__ b,
                           float* __restrict__ c, int n) {
  const int i = blockIdx.x * 256 + threadIdx.x;
  if (i < n) c[i] = a[i] + b[i];
}
__global__ void swiglu_bf16_kernel(const float* __restrict__ gate,
                                   const float* __restrict__ up,
                                   bf16_t* __restrict__ out, int n) {
  const int i = blockIdx.x * 256 + threadIdx.x;
  if (i < n) out[i] = (bf16_t)(silu1(gate[i]) * up[i]);
}

// ---------------------------------------------------------------------------
extern "C" void kernel_launch(void* const* d_in, const int* in_sizes, int n_in,
                              void* d_out, int out_size, void* d_ws, size_t ws_size,
                              hipStream_t stream) {
  const float* x      = (const float*)d_in[0];
  const float* attn_w = (const float*)d_in[1];
  const float* Wq     = (const float*)d_in[2];
  const float* Wk     = (const float*)d_in[3];
  const float* Wv     = (const float*)d_in[4];
  const float* conv_q = (const float*)d_in[5];
  const float* conv_k = (const float*)d_in[6];
  const float* conv_v = (const float*)d_in[7];
  const float* Wbeta  = (const float*)d_in[8];
  const float* out_w  = (const float*)d_in[9];
  const float* Wout   = (const float*)d_in[10];
  const float* mlp_w  = (const float*)d_in[11];
  const float* Wgate  = (const float*)d_in[12];
  const float* Wup    = (const float*)d_in[13];
  const float* Wdown  = (const float*)d_in[14];

  const size_t TH = (size_t)TSEQ * HID;        // 2,097,152
  const size_t TI = (size_t)TSEQ * INTER_DIM;  // 5,767,168
  const long H2 = (long)HID * HID;             // 4,194,304
  const long NB = (long)NHEAD * HID;           // 65,536
  const long WI = (long)INTER_DIM * HID;       // 11,534,336
  char* ws = (char*)d_ws;

  // fp32/activation arena (~91 MB) with lifetime-based aliasing
  float* qpre = (float*)ws;                        // [3*TH] (gate aliases later)
  float* kpre = qpre + TH;
  float* vpre = kpre + TH;
  float* gate = qpre;                              // alias: TI <= 3*TH
  float* qn   = (float*)(ws + 3 * TH * 4);         // [3*TH] (up aliases later)
  float* kn   = qn + TH;
  float* vn   = kn + TH;
  float* upb  = qn;                                // alias
  float* beta = (float*)(ws + 6 * TH * 4);         // [TSEQ*NHEAD]
  float* obuf = beta + (size_t)TSEQ * NHEAD;       // [TH]
  float* oproj= obuf + TH;                         // [TH] (down aliases later)
  float* x2   = oproj + TH;                        // [TH]
  bf16_t* hb  = (bf16_t*)(x2 + TH);                // [TH] bf16 (shared norm buf)
  bf16_t* act = (bf16_t*)((char*)hb + TH * 2);     // [TI] bf16
  // bf16 weight arena (~103 MB, fits the 192MB L2 for K-strip re-reads)
  bf16_t* Wq_b    = (bf16_t*)((char*)act + TI * 2);
  bf16_t* Wk_b    = Wq_b + H2;
  bf16_t* Wv_b    = Wk_b + H2;
  bf16_t* Wbeta_b = Wv_b + H2;
  bf16_t* Wout_b  = Wbeta_b + NB;
  bf16_t* Wgate_b = Wout_b + H2;
  bf16_t* Wup_b   = Wgate_b + WI;
  bf16_t* Wdown_b = Wup_b + WI;

  float* out_x = (float*)d_out;                    // [TH]
  float* out_S = out_x + TH;                       // [NHEAD*HDIM*HDIM]

  const dim3 gH((HID + BN - 1) / BN, TSEQ / BM);        // 16 x 8
  const dim3 gB((NHEAD + BN - 1) / BN, TSEQ / BM);      // 1 x 8
  const dim3 gI((INTER_DIM + BN - 1) / BN, TSEQ / BM);  // 44 x 8
  auto cgrid = [](long n) { return (unsigned)((n / 8 + 255) / 256); };

  // ---- weight fp32 -> bf16 convert pass ----
  convert_f32_bf16_kernel<<<cgrid(H2), 256, 0, stream>>>(Wq, Wq_b, H2);
  convert_f32_bf16_kernel<<<cgrid(H2), 256, 0, stream>>>(Wk, Wk_b, H2);
  convert_f32_bf16_kernel<<<cgrid(H2), 256, 0, stream>>>(Wv, Wv_b, H2);
  convert_f32_bf16_kernel<<<cgrid(NB), 256, 0, stream>>>(Wbeta, Wbeta_b, NB);
  convert_f32_bf16_kernel<<<cgrid(H2), 256, 0, stream>>>(Wout, Wout_b, H2);
  convert_f32_bf16_kernel<<<cgrid(WI), 256, 0, stream>>>(Wgate, Wgate_b, WI);
  convert_f32_bf16_kernel<<<cgrid(WI), 256, 0, stream>>>(Wup, Wup_b, WI);
  convert_f32_bf16_kernel<<<cgrid(WI), 256, 0, stream>>>(Wdown, Wdown_b, WI);

  // ---- attention sub-block ----
  rmsnorm_bf16_kernel<<<TSEQ, 256, 0, stream>>>(x, attn_w, hb);
  gemm_bf16_wmma_async_kernel<<<gH, 256, 0, stream>>>(hb, Wq_b, qpre, TSEQ, HID, HID);
  gemm_bf16_wmma_async_kernel<<<gH, 256, 0, stream>>>(hb, Wk_b, kpre, TSEQ, HID, HID);
  gemm_bf16_wmma_async_kernel<<<gH, 256, 0, stream>>>(hb, Wv_b, vpre, TSEQ, HID, HID);
  gemm_bf16_wmma_async_kernel<<<gB, 256, 0, stream>>>(hb, Wbeta_b, beta, TSEQ, NHEAD, HID);
  sigmoid_inplace_kernel<<<(TSEQ * NHEAD + 255) / 256, 256, 0, stream>>>(beta, TSEQ * NHEAD);
  conv_silu_l2_kernel<<<dim3(TSEQ, NHEAD), 64, 0, stream>>>(
      qpre, kpre, vpre, conv_q, conv_k, conv_v, qn, kn, vn);
  delta_scan_kernel<<<NHEAD, 64, 0, stream>>>(qn, kn, vn, beta, obuf, out_S);
  rmsnorm_bf16_kernel<<<TSEQ, 256, 0, stream>>>(obuf, out_w, hb);
  gemm_bf16_wmma_async_kernel<<<gH, 256, 0, stream>>>(hb, Wout_b, oproj, TSEQ, HID, HID);
  add_kernel<<<(int)((TH + 255) / 256), 256, 0, stream>>>(x, oproj, x2, (int)TH);

  // ---- MLP sub-block ----
  rmsnorm_bf16_kernel<<<TSEQ, 256, 0, stream>>>(x2, mlp_w, hb);
  gemm_bf16_wmma_async_kernel<<<gI, 256, 0, stream>>>(hb, Wgate_b, gate, TSEQ, INTER_DIM, HID);
  gemm_bf16_wmma_async_kernel<<<gI, 256, 0, stream>>>(hb, Wup_b, upb, TSEQ, INTER_DIM, HID);
  swiglu_bf16_kernel<<<(int)((TI + 255) / 256), 256, 0, stream>>>(gate, upb, act, (int)TI);
  gemm_bf16_wmma_async_kernel<<<gH, 256, 0, stream>>>(act, Wdown_b, oproj, TSEQ, HID, INTER_DIM);
  add_kernel<<<(int)((TH + 255) / 256), 256, 0, stream>>>(x2, oproj, out_x, (int)TH);
}
